// MultilaterationHead_47708496724714
// MI455X (gfx1250) — matrix operand order repaired and checked by reading, not compiled, not run on hardware
//
#include <hip/hip_runtime.h>
#include <stdint.h>

// ---------------------------------------------------------------------------
// MI455X (gfx1250) fused implementation.
// Heavy GEMMs -> v_wmma_f32_16x16x32_bf16 (fp32 accum). Pairwise MLP fully
// fused (the (B,NA,NB,H1) tensor is never materialized). W2 fragments are
// staged into LDS once per workgroup with global_load_async_to_lds_b128
// (ASYNCcnt) and consumed via ds_load. Sensitive math (softplus, 3x3 solve,
// pointnet) stays fp32 VALU.
// ---------------------------------------------------------------------------

typedef __attribute__((ext_vector_type(16))) __bf16 v16bf;
typedef __attribute__((ext_vector_type(2)))  __bf16 v2bf;
typedef __attribute__((ext_vector_type(8)))  float  v8f;

#define DEVI static __device__ __forceinline__

constexpr int BATCH = 2;
constexpr int NA    = 256;
constexpr int NB    = 256;
constexpr int EDIM  = 512;
constexpr int H1    = 300;
constexpr int H1P   = 320;   // padded hidden-1 (10 k-tiles of 32)
constexpr int KT1   = 10;    // k tiles for pair GEMM (H1P/32)
constexpr int H2    = 100;
constexpr int H2P   = 112;   // padded hidden-2 (7 n-tiles of 16)
constexpr int NT2   = 7;
constexpr int NT1   = 19;    // n tiles for proj GEMM (304/16 covers 300)
constexpr int KTE   = 16;    // k tiles for proj GEMM (512/32)

constexpr int W2FRAG_ELTS = KT1 * NT2 * 32 * 16;        // 35840 bf16 elements
constexpr size_t W2LDS_BYTES = (size_t)W2FRAG_ELTS * 2; // 71680 B (fits WGP LDS)

constexpr size_t alignup(size_t x) { return (x + 255) & ~size_t(255); }

constexpr size_t SZ_W1FRAG = (size_t)2 * KTE * NT1 * 32 * 16 * 2;   // bf16 bits
constexpr size_t SZ_W2FRAG = (size_t)W2FRAG_ELTS * 2;
constexpr size_t SZ_B1PAD  = (size_t)H1P * 4;
constexpr size_t SZ_B2PAD  = (size_t)H2P * 4;
constexpr size_t SZ_W3PAD  = (size_t)H2P * 4;
constexpr size_t SZ_PROJ   = (size_t)4 * BATCH * NA * H1P * 4;      // 4 arrays
constexpr size_t SZ_R2     = (size_t)BATCH * NA * NB * 4;

constexpr size_t OFF_W1FRAG = 0;
constexpr size_t OFF_W2FRAG = alignup(OFF_W1FRAG + SZ_W1FRAG);
constexpr size_t OFF_B1PAD  = alignup(OFF_W2FRAG + SZ_W2FRAG);
constexpr size_t OFF_B2PAD  = alignup(OFF_B1PAD  + SZ_B1PAD);
constexpr size_t OFF_W3PAD  = alignup(OFF_B2PAD  + SZ_B2PAD);
constexpr size_t OFF_PROJ   = alignup(OFF_W3PAD  + SZ_W3PAD);
constexpr size_t OFF_R2     = alignup(OFF_PROJ   + SZ_PROJ);

DEVI unsigned short f2bf_bits(float f) {
  unsigned int u = __float_as_uint(f);
  unsigned int r = u + 0x7FFFu + ((u >> 16) & 1u);   // round to nearest even
  return (unsigned short)(r >> 16);
}
DEVI v2bf pk_bf16(float lo, float hi) {
#if __has_builtin(__builtin_amdgcn_cvt_pk_bf16_f32)
  return __builtin_amdgcn_cvt_pk_bf16_f32(lo, hi);
#else
  // gfx1250 has native bf16; let the backend lower fptrunc directly.
  v2bf r;
  r[0] = (__bf16)lo;
  r[1] = (__bf16)hi;
  return r;
#endif
}
DEVI float reluf(float x)    { return fmaxf(x, 0.0f); }
DEVI float softplusf(float x){ return (x > 20.0f) ? x : log1pf(expf(x)); }
DEVI void wait_async0() {
#if __has_builtin(__builtin_amdgcn_s_wait_asynccnt)
  __builtin_amdgcn_s_wait_asynccnt(0);
#else
  asm volatile("s_wait_asynccnt 0x0" ::: "memory");
#endif
}

// ---------------------------------------------------------------------------
// Kernel 1: weight swizzle into WMMA B-fragment layout + zero-padding.
// B fragment (16-bit, 32K x 16N): lanes 0-15: n=lane, k = kt*32 + i (i<8), kt*32+8+i (i>=8);
//                                 lanes 16-31: n=lane-16, k = kt*32+8+i (i<8), kt*32+16+i (i>=8).
// ---------------------------------------------------------------------------
__global__ __launch_bounds__(256)
void prep_kernel(const float* __restrict__ W1, const float* __restrict__ b1,
                 const float* __restrict__ W2, const float* __restrict__ b2,
                 const float* __restrict__ W3, uint8_t* __restrict__ ws) {
  unsigned short* w1f = (unsigned short*)(ws + OFF_W1FRAG);
  unsigned short* w2f = (unsigned short*)(ws + OFF_W2FRAG);
  float* b1p  = (float*)(ws + OFF_B1PAD);
  float* b2p  = (float*)(ws + OFF_B2PAD);
  float* w3p  = (float*)(ws + OFF_W3PAD);
  float* proj = (float*)(ws + OFF_PROJ);

  const int NW1   = 2 * KTE * NT1 * 32 * 16;
  const int NPROJ = 4 * BATCH * NA * H1P;
  const int stride = gridDim.x * blockDim.x;
  const int tid0   = blockIdx.x * blockDim.x + threadIdx.x;

  for (int f = tid0; f < NW1; f += stride) {
    int i    = f & 15;
    int lane = (f >> 4) & 31;
    int nt   = (f >> 9) % NT1;
    int kt   = ((f >> 9) / NT1) % KTE;
    int half = f / (512 * NT1 * KTE);
    int k = kt * 32 + ((lane < 16) ? 0 : 8) + i + ((i >= 8) ? 8 : 0);
    int n = nt * 16 + (lane & 15);
    w1f[f] = (n < H1) ? f2bf_bits(W1[(size_t)(half * EDIM + k) * H1 + n])
                      : (unsigned short)0;
  }
  for (int f = tid0; f < W2FRAG_ELTS; f += stride) {
    int i    = f & 15;
    int lane = (f >> 4) & 31;
    int nt   = (f >> 9) % NT2;
    int kt   = (f >> 9) / NT2;
    int k = kt * 32 + ((lane < 16) ? 0 : 8) + i + ((i >= 8) ? 8 : 0);
    int n = nt * 16 + (lane & 15);
    w2f[f] = (k < H1 && n < H2) ? f2bf_bits(W2[(size_t)k * H2 + n])
                                : (unsigned short)0;
  }
  for (int f = tid0; f < H1P; f += stride) b1p[f] = (f < H1) ? b1[f] : 0.0f;
  for (int f = tid0; f < H2P; f += stride) {
    b2p[f] = (f < H2) ? b2[f] : 0.0f;
    w3p[f] = (f < H2) ? W3[f] : 0.0f;
  }
  for (int f = tid0; f < NPROJ; f += stride) proj[f] = 0.0f;   // pad cols 304..319
}

// ---------------------------------------------------------------------------
// Kernel 2: projection GEMMs  h = Phi @ W1half  (fp32 out, bf16 WMMA inside).
// One wave per 16x16 output tile. A fragment (16-bit 16x32):
// lanes 0-15 row m: k = kt*32 + i (i<8), kt*32+8+i (i>=8)
// lanes 16-31 row m: k = kt*32+8+i (i<8), kt*32+16+i (i>=8)
// ---------------------------------------------------------------------------
__global__ __launch_bounds__(32)
void proj_gemm_kernel(const float* __restrict__ actE,
                      const float* __restrict__ anchE,
                      uint8_t* __restrict__ ws) {
  const int lane = threadIdx.x;
  const int nt = blockIdx.x;         // 0..18
  const int mt = blockIdx.y;         // 0..15
  const int combo = blockIdx.z;      // 0..7
  const int bi   = combo >> 2;
  const int src  = (combo >> 1) & 1;
  const int half = combo & 1;

  const float* emb = src ? anchE : actE;
  const unsigned short* w1f =
      (const unsigned short*)(ws + OFF_W1FRAG) + (size_t)half * KTE * NT1 * 512;
  float* proj = (float*)(ws + OFF_PROJ);
  float* dst  = proj + (size_t)((src * 2 + half) * BATCH + bi) * NA * H1P;

  const int m  = lane & 15;
  const int hi = lane >> 4;

  v8f acc = {0.f, 0.f, 0.f, 0.f, 0.f, 0.f, 0.f, 0.f};
  for (int kt = 0; kt < KTE; ++kt) {
    float e[16];
#pragma unroll
    for (int i = 0; i < 16; ++i) {
      int k = kt * 32 + (hi ? 8 : 0) + i + ((i >= 8) ? 8 : 0);
      // Phi[m,k] = emb[b, e=k, n=m0+m]
      e[i] = emb[(size_t)(bi * EDIM + k) * NA + mt * 16 + m];
    }
    union { v16bf v; v2bf p[8]; } au;
#pragma unroll
    for (int j = 0; j < 8; ++j) au.p[j] = pk_bf16(e[2 * j], e[2 * j + 1]);

    const v16bf Bf =
        *(const v16bf*)(w1f + ((size_t)(kt * NT1 + nt) * 32 + lane) * 16);
    acc = __builtin_amdgcn_wmma_f32_16x16x32_bf16(false, au.v, false, Bf,
                                                  (short)0, acc, false, false);
  }
#pragma unroll
  for (int v = 0; v < 8; ++v) {
    int row = mt * 16 + v + (hi ? 8 : 0);
    int col = nt * 16 + m;
    dst[(size_t)row * H1P + col] = acc[v];
  }
}

// ---------------------------------------------------------------------------
// Kernel 3: fused pairwise MLP. One 256-thread block per 16x16 (a,b) tile.
// W2 fragments staged to LDS once per block via async global->LDS b128 copies
// (ASYNCcnt), then consumed by all 8 waves from LDS. Each wave handles 2
// M-tiles (pa). For each pass (v1 / v2):
//   A[row=pb,k] = relu(h_col[a0+pa,k] + h_row[b0+pb,k] + b1[k])  (bf16)
//   acc[nt]    += A @ W2frag[kt][nt]      (10 k-tiles x 7 n-tiles WMMA)
//   s[v]       += relu(acc + b2) . W3
// Then cross-lane reduce over the 16-lane C-matrix halves, softplus, write R^2.
// ---------------------------------------------------------------------------
__global__ __launch_bounds__(256)
void pair_mlp_kernel(const float* __restrict__ b3ptr, uint8_t* __restrict__ ws) {
  extern __shared__ __align__(32) unsigned char smem_raw[];
  unsigned short* w2s = (unsigned short*)smem_raw;

  const int tid  = threadIdx.x;
  const int lane = tid & 31;
  const int wv   = tid >> 5;
  const int hi   = lane >> 4;
  const int lm   = lane & 15;
  const int b0   = blockIdx.x * 16;
  const int a0   = blockIdx.y * 16;
  const int bi   = blockIdx.z;

  const unsigned short* w2f = (const unsigned short*)(ws + OFF_W2FRAG);
  const float* b1p  = (const float*)(ws + OFF_B1PAD);
  const float* b2p  = (const float*)(ws + OFF_B2PAD);
  const float* w3p  = (const float*)(ws + OFF_W3PAD);
  const float* proj = (const float*)(ws + OFF_PROJ);
  float* R2 = (float*)(ws + OFF_R2);

  // ---- async stage W2 fragments into LDS (16B granules, ASYNCcnt-tracked)
  {
    const int nchunks = W2FRAG_ELTS / 8;            // 8 bf16 = 16 bytes
    for (int c = tid; c < nchunks; c += 256) {
      unsigned int       laddr = (unsigned int)(uintptr_t)(w2s + (size_t)c * 8);
      unsigned long long gaddr = (unsigned long long)(uintptr_t)(w2f + (size_t)c * 8);
      asm volatile("global_load_async_to_lds_b128 %0, %1, off"
                   :: "v"(laddr), "v"(gaddr) : "memory");
    }
    wait_async0();
  }
  __syncthreads();

  const float* hAa = proj + (size_t)(0 * BATCH + bi) * NA * H1P;
  const float* hAb = proj + (size_t)(1 * BATCH + bi) * NA * H1P;
  const float* hBa = proj + (size_t)(2 * BATCH + bi) * NA * H1P;
  const float* hBb = proj + (size_t)(3 * BATCH + bi) * NA * H1P;

  const float b3 = b3ptr[0];
  float b2v[NT2], w3v[NT2];
#pragma unroll
  for (int nt = 0; nt < NT2; ++nt) {
    b2v[nt] = b2p[nt * 16 + lm];
    w3v[nt] = w3p[nt * 16 + lm];
  }

  const v8f vzero = {0.f, 0.f, 0.f, 0.f, 0.f, 0.f, 0.f, 0.f};

  for (int pa = wv; pa < 16; pa += 8) {
    float s[8];
#pragma unroll
    for (int v = 0; v < 8; ++v) s[v] = 0.f;

#pragma unroll
    for (int pass = 0; pass < 2; ++pass) {
      // v1: tail(hA_a[a] + hB_b[bcol]);  v2: tail(hB_a[bcol] + hA_b[a])
      const float* rowv = pass ? hBa : hBb;   // varies with pb (= A row)
      const float* colf = pass ? hAb : hAa;   // fixed row a0+pa
      const float* rp = rowv + (size_t)(b0 + lm) * H1P;
      const float* cp = colf + (size_t)(a0 + pa) * H1P;

      v8f acc[NT2];
#pragma unroll
      for (int nt = 0; nt < NT2; ++nt) acc[nt] = vzero;

      for (int kt = 0; kt < KT1; ++kt) {
        const int kb = kt * 32 + (hi ? 8 : 0);
        const float4 r0 = *(const float4*)(rp + kb);
        const float4 r1 = *(const float4*)(rp + kb + 4);
        const float4 r2 = *(const float4*)(rp + kb + 16);
        const float4 r3 = *(const float4*)(rp + kb + 20);
        const float4 c0 = *(const float4*)(cp + kb);
        const float4 c1 = *(const float4*)(cp + kb + 4);
        const float4 c2 = *(const float4*)(cp + kb + 16);
        const float4 c3 = *(const float4*)(cp + kb + 20);
        const float4 g0 = *(const float4*)(b1p + kb);
        const float4 g1 = *(const float4*)(b1p + kb + 4);
        const float4 g2 = *(const float4*)(b1p + kb + 16);
        const float4 g3 = *(const float4*)(b1p + kb + 20);

        union { v16bf v; v2bf p[8]; } au;
        au.p[0] = pk_bf16(reluf(r0.x + c0.x + g0.x), reluf(r0.y + c0.y + g0.y));
        au.p[1] = pk_bf16(reluf(r0.z + c0.z + g0.z), reluf(r0.w + c0.w + g0.w));
        au.p[2] = pk_bf16(reluf(r1.x + c1.x + g1.x), reluf(r1.y + c1.y + g1.y));
        au.p[3] = pk_bf16(reluf(r1.z + c1.z + g1.z), reluf(r1.w + c1.w + g1.w));
        au.p[4] = pk_bf16(reluf(r2.x + c2.x + g2.x), reluf(r2.y + c2.y + g2.y));
        au.p[5] = pk_bf16(reluf(r2.z + c2.z + g2.z), reluf(r2.w + c2.w + g2.w));
        au.p[6] = pk_bf16(reluf(r3.x + c3.x + g3.x), reluf(r3.y + c3.y + g3.y));
        au.p[7] = pk_bf16(reluf(r3.z + c3.z + g3.z), reluf(r3.w + c3.w + g3.w));

#pragma unroll
        for (int nt = 0; nt < NT2; ++nt) {
          const v16bf Bf =
              *(const v16bf*)(w2s + ((size_t)(kt * NT2 + nt) * 32 + lane) * 16);
          acc[nt] = __builtin_amdgcn_wmma_f32_16x16x32_bf16(
              false, au.v, false, Bf, (short)0, acc[nt], false, false);
        }
      }
      // second-layer relu + W3 dot (padded cols contribute 0 via w3v==0)
#pragma unroll
      for (int nt = 0; nt < NT2; ++nt) {
#pragma unroll
        for (int v = 0; v < 8; ++v)
          s[v] += reluf(acc[nt][v] + b2v[nt]) * w3v[nt];
      }
    }

    // reduce over the 16 lanes of each C-matrix half, softplus, write R^2
#pragma unroll
    for (int v = 0; v < 8; ++v) {
      float t = s[v];
      t += __shfl_xor(t, 1, 32);
      t += __shfl_xor(t, 2, 32);
      t += __shfl_xor(t, 4, 32);
      t += __shfl_xor(t, 8, 32);
      if (lm == 0) {
        const int pb = v + (hi ? 8 : 0);
        const float r = softplusf(0.5f * t + b3);
        R2[((size_t)(bi * NA) + a0 + pa) * NB + b0 + pb] = r * r;
      }
    }
  }
}

// ---------------------------------------------------------------------------
// Kernel 4: multilateration + flow (fp32). One block per batch.
// ---------------------------------------------------------------------------
__global__ __launch_bounds__(256)
void multilat_kernel(const float* __restrict__ actP,
                     const float* __restrict__ anchP,
                     const uint8_t* __restrict__ ws,
                     float* __restrict__ out) {
  __shared__ float pX[256], pY[256], pZ[256], sB[256];
  __shared__ float Ax[256], Ay[256], Az[256];
  __shared__ float Minv[9];
  __shared__ float qb[3];

  const int b = blockIdx.x;
  const int t = threadIdx.x;
  const float* R2 = (const float*)(ws + OFF_R2);

  const float x = anchP[(size_t)(b * 3 + 0) * NB + t];
  const float y = anchP[(size_t)(b * 3 + 1) * NB + t];
  const float z = anchP[(size_t)(b * 3 + 2) * NB + t];
  pX[t] = x; pY[t] = y; pZ[t] = z;
  sB[t] = x * x + y * y + z * z;
  __syncthreads();

  if (t == 0) {
    float sx = 0, sy = 0, sz = 0;
    for (int k = 0; k < NB; ++k) { sx += pX[k]; sy += pY[k]; sz += pZ[k]; }
    qb[0] = sx * (1.f / NB); qb[1] = sy * (1.f / NB); qb[2] = sz * (1.f / NB);
  }
  __syncthreads();

  Ax[t] = 2.f * (x - qb[0]);
  Ay[t] = 2.f * (y - qb[1]);
  Az[t] = 2.f * (z - qb[2]);
  __syncthreads();

  if (t == 0) {
    float m00 = 0, m01 = 0, m02 = 0, m11 = 0, m12 = 0, m22 = 0;
    for (int k = 0; k < NB; ++k) {
      m00 += Ax[k] * Ax[k]; m01 += Ax[k] * Ay[k]; m02 += Ax[k] * Az[k];
      m11 += Ay[k] * Ay[k]; m12 += Ay[k] * Az[k]; m22 += Az[k] * Az[k];
    }
    const float c00 = m11 * m22 - m12 * m12;
    const float c01 = m12 * m02 - m01 * m22;
    const float c02 = m01 * m12 - m11 * m02;
    const float det = m00 * c00 + m01 * c01 + m02 * c02;
    const float id  = 1.f / det;
    Minv[0] = c00 * id; Minv[1] = c01 * id; Minv[2] = c02 * id;
    Minv[3] = c01 * id; Minv[4] = (m00 * m22 - m02 * m02) * id;
    Minv[5] = (m01 * m02 - m00 * m12) * id;
    Minv[6] = c02 * id; Minv[7] = Minv[5];
    Minv[8] = (m00 * m11 - m01 * m01) * id;
  }
  __syncthreads();

  // per action point a = t
  const float* r2row = R2 + ((size_t)(b * NA) + t) * NB;
  float bb = 0, S0 = 0, S1 = 0, S2 = 0, T0 = 0, T1 = 0, T2 = 0;
  for (int k = 0; k < NB; ++k) {
    const float bv = sB[k] - r2row[k];
    bb += bv;
    S0 += Ax[k] * bv; S1 += Ay[k] * bv; S2 += Az[k] * bv;
    T0 += Ax[k];      T1 += Ay[k];      T2 += Az[k];
  }
  bb *= (1.f / NB);
  const float r0 = S0 - bb * T0;
  const float r1 = S1 - bb * T1;
  const float r2 = S2 - bb * T2;
  const float p0 = Minv[0] * r0 + Minv[1] * r1 + Minv[2] * r2;
  const float p1 = Minv[3] * r0 + Minv[4] * r1 + Minv[5] * r2;
  const float p2 = Minv[6] * r0 + Minv[7] * r1 + Minv[8] * r2;

  out[(size_t)(b * 4 + 0) * NA + t] = p0 - actP[(size_t)(b * 3 + 0) * NA + t];
  out[(size_t)(b * 4 + 1) * NA + t] = p1 - actP[(size_t)(b * 3 + 1) * NA + t];
  out[(size_t)(b * 4 + 2) * NA + t] = p2 - actP[(size_t)(b * 3 + 2) * NA + t];
}

// ---------------------------------------------------------------------------
// Kernel 5: PointNet 1x1-conv chain + head (fp32, LDS ping-pong).
// One block per (batch, 16-point tile).
// ---------------------------------------------------------------------------
DEVI void conv_layer(const float* in, int Cin, float* outp, int Cout,
                     const float* __restrict__ W, const float* __restrict__ bias,
                     int tid) {
  for (int op = tid; op < Cout * 16; op += 256) {
    const int o = op >> 4, p = op & 15;
    float acc = bias[o];
    const float* wr = W + (size_t)o * Cin;
    for (int i = 0; i < Cin; ++i) acc += wr[i] * in[i * 16 + p];
    outp[op] = fmaxf(acc, 0.0f);
  }
  __syncthreads();
}

__global__ __launch_bounds__(256)
void pointnet_kernel(const float* __restrict__ xin,
                     const float* __restrict__ W0, const float* __restrict__ bb0,
                     const float* __restrict__ W1, const float* __restrict__ bb1,
                     const float* __restrict__ W2, const float* __restrict__ bb2,
                     const float* __restrict__ W3, const float* __restrict__ bb3,
                     const float* __restrict__ W4, const float* __restrict__ bb4,
                     const float* __restrict__ hW, float* __restrict__ out) {
  __shared__ float SA[512 * 16];
  __shared__ float SB[128 * 16];
  __shared__ float SC[128 * 16];
  const int tid = threadIdx.x;
  const int n0  = blockIdx.x * 16;
  const int b   = blockIdx.y;

  for (int idx = tid; idx < 512 * 16; idx += 256) {
    const int c = idx >> 4, p = idx & 15;
    SA[idx] = xin[(size_t)(b * 512 + c) * 256 + n0 + p];
  }
  __syncthreads();

  conv_layer(SA, 512, SB,  64, W0, bb0, tid);
  conv_layer(SB,  64, SC,  64, W1, bb1, tid);
  conv_layer(SC,  64, SB,  64, W2, bb2, tid);
  conv_layer(SB,  64, SC, 128, W3, bb3, tid);
  conv_layer(SC, 128, SA, 512, W4, bb4, tid);

  for (int p = tid; p < 16; p += 256) {
    float acc = 0.f;
    for (int i = 0; i < 512; ++i) acc += hW[i] * SA[i * 16 + p];
    out[(size_t)(b * 4 + 3) * 256 + n0 + p] = acc;
  }
}

// ---------------------------------------------------------------------------
extern "C" void kernel_launch(void* const* d_in, const int* in_sizes, int n_in,
                              void* d_out, int out_size, void* d_ws, size_t ws_size,
                              hipStream_t stream) {
  const float* actE  = (const float*)d_in[0];   // (B,512,256)
  const float* anchE = (const float*)d_in[1];   // (B,512,256)
  const float* actP  = (const float*)d_in[2];   // (B,3,256)
  const float* anchP = (const float*)d_in[3];   // (B,3,256)
  const float* W1    = (const float*)d_in[4];   // (1024,300)
  const float* b1    = (const float*)d_in[5];
  const float* W2    = (const float*)d_in[6];   // (300,100)
  const float* b2    = (const float*)d_in[7];
  const float* W3    = (const float*)d_in[8];   // (100,1)
  const float* b3    = (const float*)d_in[9];
  const float* pW0 = (const float*)d_in[10]; const float* pb0 = (const float*)d_in[11];
  const float* pW1 = (const float*)d_in[12]; const float* pb1 = (const float*)d_in[13];
  const float* pW2 = (const float*)d_in[14]; const float* pb2 = (const float*)d_in[15];
  const float* pW3 = (const float*)d_in[16]; const float* pb3 = (const float*)d_in[17];
  const float* pW4 = (const float*)d_in[18]; const float* pb4 = (const float*)d_in[19];
  const float* hW  = (const float*)d_in[20];

  uint8_t* ws = (uint8_t*)d_ws;
  float*  out = (float*)d_out;

  // Allow >64KB dynamic LDS for the pair kernel (320KB available per WGP).
  // Host-side attribute set; not a stream operation, graph-capture safe.
  (void)hipFuncSetAttribute((const void*)pair_mlp_kernel,
                            hipFuncAttributeMaxDynamicSharedMemorySize,
                            (int)W2LDS_BYTES);

  prep_kernel<<<1024, 256, 0, stream>>>(W1, b1, W2, b2, W3, ws);
  proj_gemm_kernel<<<dim3(NT1, 16, 8), 32, 0, stream>>>(actE, anchE, ws);
  pair_mlp_kernel<<<dim3(16, 16, BATCH), 256, W2LDS_BYTES, stream>>>(b3, ws);
  multilat_kernel<<<BATCH, 256, 0, stream>>>(actP, anchP, ws, out);
  pointnet_kernel<<<dim3(16, BATCH), 256, 0, stream>>>(
      actE, pW0, pb0, pW1, pb1, pW2, pb2, pW3, pb3, pW4, pb4, hW, out);
}
